// IMP_7602092114048
// MI455X (gfx1250) — matrix operand order repaired and checked by reading, not compiled, not run on hardware
//
#include <hip/hip_runtime.h>

#define DIM 512
#define D_IN 4096
#define N_OBJ 2048
#define N_REL 32768
#define N_OBJ_CLS 151
#define N_REL_CLS 51
#define STEPS 2
#define EPSV 1e-5f

typedef __attribute__((ext_vector_type(16))) __bf16 bf16x16;
typedef __attribute__((ext_vector_type(8)))  float  floatx8;

__device__ __forceinline__ float sigmoidf(float x) { return 1.0f / (1.0f + __expf(-x)); }

// ---------------------------------------------------------------------------
// Generic GEMM: C[M,N] = act(A_f32[M,K] @ W_bf16[N,K]^T + bias[N])
// A is fp32 in memory, converted to bf16 in registers (RNE) per K-tile.
// W is pre-converted bf16, row-major [N][K] (K contiguous) -> B operand.
// Wave tile: 32M x 64N (2 A-frags x 4 B-frags = 8 accumulators).
// Block: 8 waves = 128M x 128N. K must be a multiple of 32.
// Emits v_wmma_f32_16x16x32_bf16 (8 per K-step, 16 b128 loads per K-step).
// ---------------------------------------------------------------------------
__global__ __launch_bounds__(256) void gemm_wmma_bf16(
    const float* __restrict__ A, const __bf16* __restrict__ W,
    const float* __restrict__ bias, float* __restrict__ C,
    int M, int N, int K, int relu)
{
    const int lane = threadIdx.x & 31;
    const int wave = threadIdx.x >> 5;
    const int wm   = wave >> 1;        // 0..3  (32 rows each)
    const int wn   = wave & 1;         // 0..1  (64 cols each)
    const int l15  = lane & 15;
    const int hi   = lane >> 4;        // 0 or 1 (lane half)

    const int row0 = blockIdx.y * 128 + wm * 32;
    const int col0 = blockIdx.x * 128 + wn * 64;

    const float* Arow[2];
#pragma unroll
    for (int r = 0; r < 2; ++r) {
        int ar = row0 + r * 16 + l15;
        if (ar >= M) ar = M - 1;
        Arow[r] = A + (size_t)ar * K;
    }
    const int kA = hi * 16;            // A lane-half K offset

    // B lane pointers: lane holds 16 contiguous K values of W row (col)
    const __bf16* Wl[4];
#pragma unroll
    for (int s = 0; s < 4; ++s) {
        int wrow = col0 + s * 16 + l15;
        if (wrow >= N) wrow = 0;       // clamp; masked at store
        Wl[s] = W + (size_t)wrow * K + hi * 16;
    }

    floatx8 acc[2][4];
#pragma unroll
    for (int r = 0; r < 2; ++r)
#pragma unroll
        for (int s = 0; s < 4; ++s)
#pragma unroll
            for (int j = 0; j < 8; ++j) acc[r][s][j] = 0.0f;

    for (int k0 = 0; k0 < K; k0 += 32) {
        // speculative prefetch two K tiles ahead (global_prefetch_b8)
        __builtin_prefetch(Arow[0] + k0 + 64, 0, 1);
        __builtin_prefetch(Arow[1] + k0 + 64, 0, 1);

        // --- A fragments: lane half 0 -> K {k0..k0+15},
        //                  lane half 1 -> K {k0+16..k0+31}
        bf16x16 af[2];
#pragma unroll
        for (int r = 0; r < 2; ++r) {
            const float4* pa = (const float4*)(Arow[r] + k0 + kA);
            float4 a0 = pa[0], a1 = pa[1];
            const float4* pb = (const float4*)(Arow[r] + k0 + 8 + kA);
            float4 a2 = pb[0], a3 = pb[1];
            af[r][0]  = (__bf16)a0.x; af[r][1]  = (__bf16)a0.y;
            af[r][2]  = (__bf16)a0.z; af[r][3]  = (__bf16)a0.w;
            af[r][4]  = (__bf16)a1.x; af[r][5]  = (__bf16)a1.y;
            af[r][6]  = (__bf16)a1.z; af[r][7]  = (__bf16)a1.w;
            af[r][8]  = (__bf16)a2.x; af[r][9]  = (__bf16)a2.y;
            af[r][10] = (__bf16)a2.z; af[r][11] = (__bf16)a2.w;
            af[r][12] = (__bf16)a3.x; af[r][13] = (__bf16)a3.y;
            af[r][14] = (__bf16)a3.z; af[r][15] = (__bf16)a3.w;
        }

        // --- B fragments + WMMA (each B fragment reused by both A frags)
#pragma unroll
        for (int s = 0; s < 4; ++s) {
            bf16x16 bf = *(const bf16x16*)(Wl[s] + k0);
            acc[0][s] = __builtin_amdgcn_wmma_f32_16x16x32_bf16(
                false, af[0], false, bf, (short)0, acc[0][s], false, false);
            acc[1][s] = __builtin_amdgcn_wmma_f32_16x16x32_bf16(
                false, af[1], false, bf, (short)0, acc[1][s], false, false);
        }
    }

    // --- epilogue: C/D layout: lane -> N=l15, vgpr j -> M = j + hi*8
#pragma unroll
    for (int r = 0; r < 2; ++r) {
#pragma unroll
        for (int s = 0; s < 4; ++s) {
            int n = col0 + s * 16 + l15;
            if (n < N) {
                float bv = bias ? bias[n] : 0.0f;
#pragma unroll
                for (int j = 0; j < 8; ++j) {
                    int m = row0 + r * 16 + j + hi * 8;
                    if (m < M) {
                        float v = acc[r][s][j] + bv;
                        if (relu) v = fmaxf(v, 0.0f);
                        C[(size_t)m * N + n] = v;
                    }
                }
            }
        }
    }
}

// ---------------------------------------------------------------------------
// fp32 -> bf16 weight conversion
// ---------------------------------------------------------------------------
__global__ void f32_to_bf16_kernel(const float* __restrict__ in, __bf16* __restrict__ out, int n)
{
    int i = blockIdx.x * blockDim.x + threadIdx.x;
    if (i < n) out[i] = (__bf16)in[i];
}

// ---------------------------------------------------------------------------
// counts: cnt_s[subj]++, cnt_o[obj]++
// ---------------------------------------------------------------------------
__global__ void counts_kernel(const int* __restrict__ rel, float* cs, float* co)
{
    int e = blockIdx.x * blockDim.x + threadIdx.x;
    if (e < N_REL) {
        atomicAdd(&cs[rel[2 * e]], 1.0f);
        atomicAdd(&co[rel[2 * e + 1]], 1.0f);
    }
}

// ---------------------------------------------------------------------------
// gates: one wave per edge; 8 dot products of length DIM, wave32 reduction
// gates[e*4+0]=sn, +1=on, +2=se, +3=oe
// ---------------------------------------------------------------------------
__global__ __launch_bounds__(256) void gates_kernel(
    const float* __restrict__ h_obj, const float* __restrict__ h_edge,
    const int* __restrict__ rel,
    const float* __restrict__ wsn, const float* __restrict__ bsn,
    const float* __restrict__ won, const float* __restrict__ bon,
    const float* __restrict__ wse, const float* __restrict__ bse,
    const float* __restrict__ woe, const float* __restrict__ boe,
    float* __restrict__ gates)
{
    int e = blockIdx.x * (blockDim.x >> 5) + (threadIdx.x >> 5);
    int lane = threadIdx.x & 31;
    if (e >= N_REL) return;
    int s = rel[2 * e], o = rel[2 * e + 1];
    const float* hs = h_obj + (size_t)s * DIM;
    const float* ho = h_obj + (size_t)o * DIM;
    const float* he = h_edge + (size_t)e * DIM;

    float a_sn = 0, a_on = 0, a_se = 0, a_oe = 0;
    float b_sn = 0, b_on = 0, b_se = 0, b_oe = 0;
    for (int d = lane; d < DIM; d += 32) {
        float vs = hs[d], vo = ho[d], ve = he[d];
        a_sn += vs * wsn[d];  b_sn += ve * wsn[DIM + d];
        a_on += vo * won[d];  b_on += ve * won[DIM + d];
        a_se += vs * wse[d];  b_se += ve * wse[DIM + d];
        a_oe += vo * woe[d];  b_oe += ve * woe[DIM + d];
    }
    for (int off = 16; off > 0; off >>= 1) {
        a_sn += __shfl_down(a_sn, off, 32);  b_sn += __shfl_down(b_sn, off, 32);
        a_on += __shfl_down(a_on, off, 32);  b_on += __shfl_down(b_on, off, 32);
        a_se += __shfl_down(a_se, off, 32);  b_se += __shfl_down(b_se, off, 32);
        a_oe += __shfl_down(a_oe, off, 32);  b_oe += __shfl_down(b_oe, off, 32);
    }
    if (lane == 0) {
        gates[4 * e + 0] = sigmoidf(a_sn + b_sn + bsn[0]);
        gates[4 * e + 1] = sigmoidf(a_on + b_on + bon[0]);
        gates[4 * e + 2] = sigmoidf(a_se + b_se + bse[0]);
        gates[4 * e + 3] = sigmoidf(a_oe + b_oe + boe[0]);
    }
}

// ---------------------------------------------------------------------------
// scatter node messages: accA[s] += g_sn*h_edge[e], accB[o] += g_on*h_edge[e]
// ---------------------------------------------------------------------------
__global__ __launch_bounds__(256) void scatter_kernel(
    const float* __restrict__ h_edge, const int* __restrict__ rel,
    const float* __restrict__ gates, float* __restrict__ accA, float* __restrict__ accB)
{
    int e = blockIdx.x * (blockDim.x >> 5) + (threadIdx.x >> 5);
    int lane = threadIdx.x & 31;
    if (e >= N_REL) return;
    int s = rel[2 * e], o = rel[2 * e + 1];
    float gsn = gates[4 * e + 0], gon = gates[4 * e + 1];
    const float* he = h_edge + (size_t)e * DIM;
    float* As = accA + (size_t)s * DIM;
    float* Ao = accB + (size_t)o * DIM;
    for (int d = lane; d < DIM; d += 32) {
        float v = he[d];
        atomicAdd(&As[d], gsn * v);
        atomicAdd(&Ao[d], gon * v);
    }
}

// node_message = 0.5*(accA/(cnt_s+eps) + accB/(cnt_o+eps))
__global__ void combine_msg_kernel(const float* accA, const float* accB,
                                   const float* cs, const float* co, float* out)
{
    int i = blockIdx.x * blockDim.x + threadIdx.x;
    if (i >= N_OBJ * DIM) return;
    int obj = i / DIM;
    out[i] = 0.5f * (accA[i] / (cs[obj] + EPSV) + accB[i] / (co[obj] + EPSV));
}

// edge_message = 0.5*(g_se*h_obj[s] + g_oe*h_obj[o])
__global__ void edge_msg_kernel(const float* __restrict__ h_obj, const int* __restrict__ rel,
                                const float* __restrict__ gates, float* __restrict__ out)
{
    int idx = blockIdx.x * blockDim.x + threadIdx.x;
    if (idx >= N_REL * DIM) return;
    int e = idx / DIM, d = idx - e * DIM;
    int s = rel[2 * e], o = rel[2 * e + 1];
    out[idx] = 0.5f * (gates[4 * e + 2] * h_obj[(size_t)s * DIM + d] +
                       gates[4 * e + 3] * h_obj[(size_t)o * DIM + d]);
}

// GRU elementwise combine (gi, gh are [M, 3*DIM]); out may alias h
__global__ void gru_elem_kernel(const float* __restrict__ gi, const float* __restrict__ gh,
                                const float* __restrict__ h, float* __restrict__ out, int M)
{
    int idx = blockIdx.x * blockDim.x + threadIdx.x;
    if (idx >= M * DIM) return;
    int m = idx / DIM, d = idx - m * DIM;
    size_t base = (size_t)m * (3 * DIM) + d;
    float ir = gi[base], iz = gi[base + DIM], in = gi[base + 2 * DIM];
    float hr = gh[base], hz = gh[base + DIM], hn = gh[base + 2 * DIM];
    float r = sigmoidf(ir + hr);
    float z = sigmoidf(iz + hz);
    float n = tanhf(in + r * hn);
    out[idx] = (1.0f - z) * n + z * h[idx];
}

// argmax over logits[:,1:] + 1, written as float
__global__ __launch_bounds__(256) void argmax_kernel(const float* __restrict__ logits,
                                                     float* __restrict__ labels)
{
    int r = blockIdx.x * (blockDim.x >> 5) + (threadIdx.x >> 5);
    int lane = threadIdx.x & 31;
    if (r >= N_OBJ) return;
    const float* row = logits + (size_t)r * N_OBJ_CLS;
    float best = -3.4e38f; int bi = N_OBJ_CLS;
    for (int c = 1 + lane; c < N_OBJ_CLS; c += 32) {
        float v = row[c];
        if (v > best || (v == best && c < bi)) { best = v; bi = c; }
    }
    for (int off = 16; off > 0; off >>= 1) {
        float v2 = __shfl_down(best, off, 32);
        int   i2 = __shfl_down(bi, off, 32);
        if (v2 > best || (v2 == best && i2 < bi)) { best = v2; bi = i2; }
    }
    if (lane == 0) labels[r] = (float)bi;
}

__global__ void i32_to_f32_kernel(const int* __restrict__ in, float* __restrict__ out, int n)
{
    int i = blockIdx.x * blockDim.x + threadIdx.x;
    if (i < n) out[i] = (float)in[i];
}

// ---------------------------------------------------------------------------
extern "C" void kernel_launch(void* const* d_in, const int* in_sizes, int n_in,
                              void* d_out, int out_size, void* d_ws, size_t ws_size,
                              hipStream_t stream)
{
    (void)in_sizes; (void)n_in; (void)out_size; (void)ws_size;

    const float* x_obj   = (const float*)d_in[0];
    const float* x_pred  = (const float*)d_in[1];
    const int*   rel     = (const int*)d_in[2];
    const float* oe_w1 = (const float*)d_in[3],  *oe_b1 = (const float*)d_in[4];
    const float* oe_w2 = (const float*)d_in[5],  *oe_b2 = (const float*)d_in[6];
    const float* pe_w1 = (const float*)d_in[7],  *pe_b1 = (const float*)d_in[8];
    const float* pe_w2 = (const float*)d_in[9],  *pe_b2 = (const float*)d_in[10];
    const float* g_sn_w = (const float*)d_in[11], *g_sn_b = (const float*)d_in[12];
    const float* g_on_w = (const float*)d_in[13], *g_on_b = (const float*)d_in[14];
    const float* g_se_w = (const float*)d_in[15], *g_se_b = (const float*)d_in[16];
    const float* g_oe_w = (const float*)d_in[17], *g_oe_b = (const float*)d_in[18];
    const float* ngru_wih = (const float*)d_in[19], *ngru_whh = (const float*)d_in[20];
    const float* ngru_bih = (const float*)d_in[21], *ngru_bhh = (const float*)d_in[22];
    const float* egru_wih = (const float*)d_in[23], *egru_whh = (const float*)d_in[24];
    const float* egru_bih = (const float*)d_in[25], *egru_bhh = (const float*)d_in[26];
    const float* op_w = (const float*)d_in[27], *op_b = (const float*)d_in[28];
    const float* pp_w = (const float*)d_in[29], *pp_b = (const float*)d_in[30];

    // ---- workspace layout (256B aligned bump allocator) ----
    char* wp = (char*)d_ws;
    auto alloc = [&](size_t bytes) -> char* {
        char* r = wp; wp += (bytes + 255) & ~(size_t)255; return r;
    };
    __bf16* wb_oe1  = (__bf16*)alloc((size_t)DIM * D_IN * 2);
    __bf16* wb_oe2  = (__bf16*)alloc((size_t)DIM * DIM * 2);
    __bf16* wb_pe1  = (__bf16*)alloc((size_t)DIM * D_IN * 2);
    __bf16* wb_pe2  = (__bf16*)alloc((size_t)DIM * DIM * 2);
    __bf16* wb_nwih = (__bf16*)alloc((size_t)3 * DIM * DIM * 2);
    __bf16* wb_nwhh = (__bf16*)alloc((size_t)3 * DIM * DIM * 2);
    __bf16* wb_ewih = (__bf16*)alloc((size_t)3 * DIM * DIM * 2);
    __bf16* wb_ewhh = (__bf16*)alloc((size_t)3 * DIM * DIM * 2);
    __bf16* wb_op   = (__bf16*)alloc((size_t)N_OBJ_CLS * DIM * 2);
    __bf16* wb_pp   = (__bf16*)alloc((size_t)N_REL_CLS * DIM * 2);
    float* hA      = (float*)alloc((size_t)N_OBJ * DIM * 4);
    float* hB      = (float*)alloc((size_t)N_OBJ * DIM * 4);
    float* h_edge  = (float*)alloc((size_t)N_REL * DIM * 4);
    float* tmp     = (float*)alloc((size_t)N_REL * DIM * 4);   // MLP hidden / edge msg
    float* nodemsg = (float*)alloc((size_t)N_OBJ * DIM * 4);
    float* accA    = (float*)alloc((size_t)N_OBJ * DIM * 4);
    float* accB    = (float*)alloc((size_t)N_OBJ * DIM * 4);
    float* cnt_s   = (float*)alloc((size_t)N_OBJ * 4);
    float* cnt_o   = (float*)alloc((size_t)N_OBJ * 4);
    float* gates   = (float*)alloc((size_t)N_REL * 4 * 4);
    float* gi      = (float*)alloc((size_t)N_REL * 3 * DIM * 4);
    float* gh      = (float*)alloc((size_t)N_REL * 3 * DIM * 4);

    auto cvt = [&](const float* src, __bf16* dst, int n) {
        f32_to_bf16_kernel<<<(n + 255) / 256, 256, 0, stream>>>(src, dst, n);
    };
    auto gemm = [&](const float* A, const __bf16* W, const float* bias, float* C,
                    int M, int N, int K, int relu) {
        dim3 g((N + 127) / 128, (M + 127) / 128);
        gemm_wmma_bf16<<<g, 256, 0, stream>>>(A, W, bias, C, M, N, K, relu);
    };

    // ---- weight conversions (per-launch; deterministic) ----
    cvt(oe_w1, wb_oe1, DIM * D_IN);      cvt(oe_w2, wb_oe2, DIM * DIM);
    cvt(pe_w1, wb_pe1, DIM * D_IN);      cvt(pe_w2, wb_pe2, DIM * DIM);
    cvt(ngru_wih, wb_nwih, 3 * DIM * DIM); cvt(ngru_whh, wb_nwhh, 3 * DIM * DIM);
    cvt(egru_wih, wb_ewih, 3 * DIM * DIM); cvt(egru_whh, wb_ewhh, 3 * DIM * DIM);
    cvt(op_w, wb_op, N_OBJ_CLS * DIM);   cvt(pp_w, wb_pp, N_REL_CLS * DIM);

    // ---- encoders ----
    gemm(x_obj,  wb_oe1, oe_b1, tmp,   N_OBJ, DIM, D_IN, 1);
    gemm(tmp,    wb_oe2, oe_b2, hA,    N_OBJ, DIM, DIM, 0);
    gemm(x_pred, wb_pe1, pe_b1, tmp,   N_REL, DIM, D_IN, 1);
    gemm(tmp,    wb_pe2, pe_b2, h_edge, N_REL, DIM, DIM, 0);

    // ---- counts ----
    hipMemsetAsync(cnt_s, 0, (size_t)N_OBJ * 4, stream);
    hipMemsetAsync(cnt_o, 0, (size_t)N_OBJ * 4, stream);
    counts_kernel<<<(N_REL + 255) / 256, 256, 0, stream>>>(rel, cnt_s, cnt_o);

    float* h_cur = hA;
    float* h_new = hB;
    const int edgeWaveBlocks = (N_REL + 7) / 8;   // 8 waves per 256-thread block

    for (int step = 0; step < STEPS; ++step) {
        gates_kernel<<<edgeWaveBlocks, 256, 0, stream>>>(
            h_cur, h_edge, rel, g_sn_w, g_sn_b, g_on_w, g_on_b,
            g_se_w, g_se_b, g_oe_w, g_oe_b, gates);

        hipMemsetAsync(accA, 0, (size_t)N_OBJ * DIM * 4, stream);
        hipMemsetAsync(accB, 0, (size_t)N_OBJ * DIM * 4, stream);
        scatter_kernel<<<edgeWaveBlocks, 256, 0, stream>>>(h_edge, rel, gates, accA, accB);
        combine_msg_kernel<<<(N_OBJ * DIM + 255) / 256, 256, 0, stream>>>(
            accA, accB, cnt_s, cnt_o, nodemsg);

        // node GRU
        gemm(nodemsg, wb_nwih, ngru_bih, gi, N_OBJ, 3 * DIM, DIM, 0);
        gemm(h_cur,   wb_nwhh, ngru_bhh, gh, N_OBJ, 3 * DIM, DIM, 0);
        gru_elem_kernel<<<(N_OBJ * DIM + 255) / 256, 256, 0, stream>>>(
            gi, gh, h_cur, h_new, N_OBJ);

        // edge message (uses OLD h_obj) + edge GRU (in-place h_edge update)
        edge_msg_kernel<<<(N_REL * DIM + 255) / 256, 256, 0, stream>>>(
            h_cur, rel, gates, tmp);
        gemm(tmp,    wb_ewih, egru_bih, gi, N_REL, 3 * DIM, DIM, 0);
        gemm(h_edge, wb_ewhh, egru_bhh, gh, N_REL, 3 * DIM, DIM, 0);
        gru_elem_kernel<<<(N_REL * DIM + 255) / 256, 256, 0, stream>>>(
            gi, gh, h_edge, h_edge, N_REL);

        float* t = h_cur; h_cur = h_new; h_new = t;
    }

    // ---- outputs: [h_obj, h_edge, obj_logits, pred_logits, labels, rel_inds] ----
    float* out = (float*)d_out;
    float* out_hobj    = out;
    float* out_hedge   = out + (size_t)N_OBJ * DIM;
    float* out_ologits = out_hedge + (size_t)N_REL * DIM;
    float* out_plogits = out_ologits + (size_t)N_OBJ * N_OBJ_CLS;
    float* out_labels  = out_plogits + (size_t)N_REL * N_REL_CLS;
    float* out_rel     = out_labels + N_OBJ;

    hipMemcpyAsync(out_hobj, h_cur, (size_t)N_OBJ * DIM * 4,
                   hipMemcpyDeviceToDevice, stream);
    hipMemcpyAsync(out_hedge, h_edge, (size_t)N_REL * DIM * 4,
                   hipMemcpyDeviceToDevice, stream);

    gemm(h_cur,  wb_op, op_b, out_ologits, N_OBJ, N_OBJ_CLS, DIM, 0);
    gemm(h_edge, wb_pp, pp_b, out_plogits, N_REL, N_REL_CLS, DIM, 0);

    argmax_kernel<<<(N_OBJ + 7) / 8, 256, 0, stream>>>(out_ologits, out_labels);
    i32_to_f32_kernel<<<(N_REL * 2 + 255) / 256, 256, 0, stream>>>(rel, out_rel, N_REL * 2);
}
